// Decoder_60627758350737
// MI455X (gfx1250) — compile-verified, matrix operand order checked by reading, not run on hardware
//
#include <hip/hip_runtime.h>
#include <hip/hip_bf16.h>

// ---------------------------------------------------------------------------
// MI455X / gfx1250. wave32, WMMA bf16 16x16x32 (fp32 accum), async LDS copies.
// N=2048 nodes, K=48 neighbors (= 3 row-tiles of 16), NF=384, MIN=1536, L=3.
// ---------------------------------------------------------------------------

typedef __bf16 bf16;
typedef __attribute__((ext_vector_type(16))) __bf16 v16bf;
typedef __attribute__((ext_vector_type(8)))  __bf16 bf16x8;
typedef __attribute__((ext_vector_type(8)))  float  v8f;

#define NN   2048
#define KK   48
#define NF   384
#define MINW 1536
#define SCALE_INV (1.0f/30.0f)
#define LN_EPS 1e-5f

__device__ __forceinline__ float gelu_f(float x) {
    return 0.5f * x * (1.0f + erff(x * 0.70710678118654752f));
}

// --- gfx1250 async global->LDS copy (VGLOBAL op 98), tracked by ASYNCcnt ---
// vdst = 32-bit LDS byte offset (generic shared addr low 32 bits, per the
// LDS aperture rule), vaddr = 64-bit global address VGPR pair.
__device__ __forceinline__ void g2l_async_b128(const void* gsrc, void* ldst) {
    unsigned long long ga = (unsigned long long)gsrc;
    unsigned int       la = (unsigned int)(unsigned long long)ldst;
    asm volatile("global_load_async_to_lds_b128 %0, %1, off"
                 :: "v"(la), "v"(ga) : "memory");
}
__device__ __forceinline__ void wait_async_all() {
    asm volatile("s_wait_asynccnt 0" ::: "memory");
}

// Cooperative async stage of `bytes` (multiple of 16) from global to LDS.
__device__ __forceinline__ void stage_async(const void* g, void* l, int bytes) {
    for (int o = threadIdx.x * 16; o < bytes; o += 256 * 16)
        g2l_async_b128((const char*)g + o, (char*)l + o);
}

// Load a 16x32 bf16 WMMA A/B fragment from a row-major [rows][ld] buffer.
// CDNA5 layout: lane L holds row (row0 + L%16); lanes<16 take K k0..k0+7 and
// k0+16..k0+23, lanes>=16 take k0+8..15 and k0+24..31. Two b128 loads/lane.
__device__ __forceinline__ v16bf frag16(const bf16* base, int row0, int ld, int k0) {
    const int lane = threadIdx.x & 31;
    const bf16* p = base + (size_t)(row0 + (lane & 15)) * ld + k0 + ((lane >> 4) << 3);
    union { v16bf v; bf16x8 h[2]; } u;
    u.h[0] = *(const bf16x8*)(p);
    u.h[1] = *(const bf16x8*)(p + 16);
    return u.v;
}

__device__ __forceinline__ v8f wmma_bf16(v16bf a, v16bf b, v8f c) {
    return __builtin_amdgcn_wmma_f32_16x16x32_bf16(false, a, false, b, (short)0, c, false, false);
}

// ----------------------------- utility kernels -----------------------------

__global__ void k_cvt_bf16(const float* __restrict__ s, bf16* __restrict__ d, int n) {
    for (int i = blockIdx.x * blockDim.x + threadIdx.x; i < n; i += gridDim.x * blockDim.x)
        d[i] = (bf16)s[i];
}

__global__ void k_copy_f32(const float* __restrict__ s, float* __restrict__ d, int n) {
    for (int i = blockIdx.x * blockDim.x + threadIdx.x; i < n; i += gridDim.x * blockDim.x)
        d[i] = s[i];
}

// dst[c][k] = (bf16) src[k][c]   (src: fp32 [Kdim][Cdim] row-major)
__global__ void k_transpose_cvt(const float* __restrict__ src, bf16* __restrict__ dst,
                                int Kdim, int Cdim, int total) {
    for (int e = blockIdx.x * blockDim.x + threadIdx.x; e < total; e += gridDim.x * blockDim.x) {
        int c = e / Kdim;
        int k = e - c * Kdim;
        dst[e] = (bf16)src[(size_t)k * Cdim + c];
    }
}

// ------------------- 16-row GEMM: out_f32 = A_bf16 @ Wt^T (+bias) ----------
// A: [rows][384] bf16, Wt: [384 out-cols][384] bf16, out: [rows][384] fp32.
__global__ void __launch_bounds__(256)
k_gemm16_f32(const bf16* __restrict__ A, const bf16* __restrict__ Wt,
             const float* __restrict__ bias, float* __restrict__ out) {
    __shared__ bf16 sX[16 * NF];                       // 12 KB A-stripe
    const int row0 = blockIdx.x * 16;
    const int tid = threadIdx.x;
    stage_async(A + (size_t)row0 * NF, sX, 16 * NF * 2);
    wait_async_all();
    __syncthreads();

    const int w = tid >> 5, lane = tid & 31;
    for (int nt = w; nt < 24; nt += 8) {
        v8f acc = {};
        #pragma unroll 4
        for (int ks = 0; ks < 12; ++ks)
            acc = wmma_bf16(frag16(sX, 0, NF, ks * 32),
                            frag16(Wt, nt * 16, NF, ks * 32), acc);
        const int col  = nt * 16 + (lane & 15);
        const int rowb = (lane >> 4) << 3;
        const float bb = bias ? bias[col] : 0.0f;
        #pragma unroll
        for (int r = 0; r < 8; ++r)
            out[(size_t)(row0 + rowb + r) * NF + col] = acc[r] + bb;
    }
}

// ------------------------- fused message kernel ----------------------------
// One workgroup per node i. Rows = 48 neighbors (3 row-tiles).
//   h0  = gelu(E @ We + P[i] + Q[idx[i,k]])        (P carries x@Wx + b0)
//   h1  = gelu(h0 @ W1 + b1)
//   msg = h1 @ W2 + b2
//   x1  = LN1(x + sum_k msg / 30)  -> x1_f32, x1_bf16
// Each wave owns 3 col-tiles; per col-tile it keeps 3 row-tile accumulators
// so every weight fragment is loaded once and reused 3x.
__global__ void __launch_bounds__(256)
k_message(const bf16* __restrict__ E16, const int* __restrict__ nbr,
          const float* __restrict__ P, const float* __restrict__ Q,
          const bf16* __restrict__ WeT, const bf16* __restrict__ W1T,
          const bf16* __restrict__ W2T,
          const float* __restrict__ b1, const float* __restrict__ b2,
          const float* __restrict__ ln1w, const float* __restrict__ ln1b,
          const float* __restrict__ xf,
          float* __restrict__ x1f, bf16* __restrict__ x1b) {
    __shared__ bf16  sA[KK * NF];     // 36 KB : E, then h1
    __shared__ bf16  sB[KK * NF];     // 36 KB : h0
    __shared__ float sF[KK * NF];     // 72 KB : add-term, then msg (fp32)
    __shared__ float sRow[NF];
    __shared__ float red1[256], red2[256];
    __shared__ int   sIdx[KK];

    const int node = blockIdx.x;
    const int tid  = threadIdx.x;
    const int w = tid >> 5, lane = tid & 31;

    // Async-stage edge features (bf16); fetch neighbor indices.
    stage_async(E16 + (size_t)node * KK * NF, sA, KK * NF * 2);
    if (tid < KK) sIdx[tid] = nbr[node * KK + tid];
    wait_async_all();
    __syncthreads();

    // Per-row additive term: P[node] + Q[neighbor]  (fp32).
    for (int e = tid; e < KK * NF; e += 256) {
        int k = e / NF, c = e - k * NF;
        sF[e] = P[(size_t)node * NF + c] + Q[(size_t)sIdx[k] * NF + c];
    }
    __syncthreads();

    // GEMM1: h0 = gelu(E @ We + add).
    for (int nt = w; nt < 24; nt += 8) {
        v8f a0 = {}, a1 = {}, a2 = {};
        #pragma unroll 4
        for (int ks = 0; ks < 12; ++ks) {
            v16bf b = frag16(WeT, nt * 16, NF, ks * 32);
            a0 = wmma_bf16(frag16(sA,  0, NF, ks * 32), b, a0);
            a1 = wmma_bf16(frag16(sA, 16, NF, ks * 32), b, a1);
            a2 = wmma_bf16(frag16(sA, 32, NF, ks * 32), b, a2);
        }
        const int col  = nt * 16 + (lane & 15);
        const int rowb = (lane >> 4) << 3;
        #pragma unroll
        for (int r = 0; r < 8; ++r) {
            int e0 = (rowb + r) * NF + col;
            int e1 = e0 + 16 * NF;
            int e2 = e0 + 32 * NF;
            sB[e0] = (bf16)gelu_f(a0[r] + sF[e0]);
            sB[e1] = (bf16)gelu_f(a1[r] + sF[e1]);
            sB[e2] = (bf16)gelu_f(a2[r] + sF[e2]);
        }
    }
    __syncthreads();

    // GEMM2: h1 = gelu(h0 @ W1 + b1)  (writes over sA).
    for (int nt = w; nt < 24; nt += 8) {
        v8f a0 = {}, a1 = {}, a2 = {};
        #pragma unroll 4
        for (int ks = 0; ks < 12; ++ks) {
            v16bf b = frag16(W1T, nt * 16, NF, ks * 32);
            a0 = wmma_bf16(frag16(sB,  0, NF, ks * 32), b, a0);
            a1 = wmma_bf16(frag16(sB, 16, NF, ks * 32), b, a1);
            a2 = wmma_bf16(frag16(sB, 32, NF, ks * 32), b, a2);
        }
        const int col  = nt * 16 + (lane & 15);
        const int rowb = (lane >> 4) << 3;
        const float bb = b1[col];
        #pragma unroll
        for (int r = 0; r < 8; ++r) {
            int e0 = (rowb + r) * NF + col;
            sA[e0]           = (bf16)gelu_f(a0[r] + bb);
            sA[e0 + 16 * NF] = (bf16)gelu_f(a1[r] + bb);
            sA[e0 + 32 * NF] = (bf16)gelu_f(a2[r] + bb);
        }
    }
    __syncthreads();

    // GEMM3: msg = h1 @ W2 + b2  (fp32 into sF).
    for (int nt = w; nt < 24; nt += 8) {
        v8f a0 = {}, a1 = {}, a2 = {};
        #pragma unroll 4
        for (int ks = 0; ks < 12; ++ks) {
            v16bf b = frag16(W2T, nt * 16, NF, ks * 32);
            a0 = wmma_bf16(frag16(sA,  0, NF, ks * 32), b, a0);
            a1 = wmma_bf16(frag16(sA, 16, NF, ks * 32), b, a1);
            a2 = wmma_bf16(frag16(sA, 32, NF, ks * 32), b, a2);
        }
        const int col  = nt * 16 + (lane & 15);
        const int rowb = (lane >> 4) << 3;
        const float bb = b2[col];
        #pragma unroll
        for (int r = 0; r < 8; ++r) {
            int e0 = (rowb + r) * NF + col;
            sF[e0]           = a0[r] + bb;
            sF[e0 + 16 * NF] = a1[r] + bb;
            sF[e0 + 32 * NF] = a2[r] + bb;
        }
    }
    __syncthreads();

    // Neighbor reduction (fp32) + residual.
    for (int cc = tid; cc < NF; cc += 256) {
        float s = 0.0f;
        #pragma unroll 8
        for (int k = 0; k < KK; ++k) s += sF[k * NF + cc];
        sRow[cc] = s * SCALE_INV + xf[(size_t)node * NF + cc];
    }
    __syncthreads();

    // LayerNorm 1 across 384 features.
    float s1 = 0.0f, s2 = 0.0f;
    for (int cc = tid; cc < NF; cc += 256) { float v = sRow[cc]; s1 += v; s2 += v * v; }
    red1[tid] = s1; red2[tid] = s2;
    __syncthreads();
    for (int off = 128; off > 0; off >>= 1) {
        if (tid < off) { red1[tid] += red1[tid + off]; red2[tid] += red2[tid + off]; }
        __syncthreads();
    }
    const float mean = red1[0] * (1.0f / NF);
    const float var  = red2[0] * (1.0f / NF) - mean * mean;
    const float rs   = rsqrtf(var + LN_EPS);
    for (int cc = tid; cc < NF; cc += 256) {
        float v = (sRow[cc] - mean) * rs * ln1w[cc] + ln1b[cc];
        x1f[(size_t)node * NF + cc] = v;
        x1b[(size_t)node * NF + cc] = (bf16)v;
    }
}

// --------------------- dense MLP stage 1: 384 -> 1536 ----------------------
__global__ void __launch_bounds__(256)
k_dense1(const bf16* __restrict__ x1b, const bf16* __restrict__ W0dT,
         const float* __restrict__ b0d, bf16* __restrict__ H) {
    __shared__ bf16 sX[16 * NF];                       // 12 KB A-stripe
    const int row0 = blockIdx.x * 16;
    const int tid = threadIdx.x;
    stage_async(x1b + (size_t)row0 * NF, sX, 16 * NF * 2);
    wait_async_all();
    __syncthreads();

    const int w = tid >> 5, lane = tid & 31;
    for (int nt = w; nt < 96; nt += 8) {
        v8f acc = {};
        #pragma unroll 4
        for (int ks = 0; ks < 12; ++ks)
            acc = wmma_bf16(frag16(sX, 0, NF, ks * 32),
                            frag16(W0dT, nt * 16, NF, ks * 32), acc);
        const int col = nt * 16 + (lane & 15);
        const int rowb = (lane >> 4) << 3;
        const float bb = b0d[col];
        #pragma unroll
        for (int r = 0; r < 8; ++r)
            H[(size_t)(row0 + rowb + r) * MINW + col] = (bf16)gelu_f(acc[r] + bb);
    }
}

// ------------- dense MLP stage 2: 1536 -> 384, fused LN2 + mask ------------
__global__ void __launch_bounds__(256)
k_dense2(const bf16* __restrict__ H, const bf16* __restrict__ W1dT,
         const float* __restrict__ b1d, const float* __restrict__ x1f,
         const float* __restrict__ maskp,
         const float* __restrict__ ln2w, const float* __restrict__ ln2b,
         float* __restrict__ outF, bf16* __restrict__ outB) {
    __shared__ bf16  sH[16 * MINW];                    // 48 KB A-stripe
    __shared__ float sD[16 * NF];                      // 24 KB result tile
    __shared__ float red1[256], red2[256];
    const int row0 = blockIdx.x * 16;
    const int tid = threadIdx.x;
    stage_async(H + (size_t)row0 * MINW, sH, 16 * MINW * 2);
    wait_async_all();
    __syncthreads();

    const int w = tid >> 5, lane = tid & 31;
    for (int nt = w; nt < 24; nt += 8) {
        v8f acc = {};
        #pragma unroll 4
        for (int ks = 0; ks < 48; ++ks)
            acc = wmma_bf16(frag16(sH, 0, MINW, ks * 32),
                            frag16(W1dT, nt * 16, MINW, ks * 32), acc);
        const int col = nt * 16 + (lane & 15);
        const int rowb = (lane >> 4) << 3;
        const float bb = b1d[col];
        #pragma unroll
        for (int r = 0; r < 8; ++r)
            sD[(rowb + r) * NF + col] = acc[r] + bb;
    }
    __syncthreads();

    for (int r = 0; r < 16; ++r) {
        const int grow = row0 + r;
        float s1 = 0.0f, s2 = 0.0f;
        for (int cc = tid; cc < NF; cc += 256) {
            float v = sD[r * NF + cc] + x1f[(size_t)grow * NF + cc];
            sD[r * NF + cc] = v;
            s1 += v; s2 += v * v;
        }
        red1[tid] = s1; red2[tid] = s2;
        __syncthreads();
        for (int off = 128; off > 0; off >>= 1) {
            if (tid < off) { red1[tid] += red1[tid + off]; red2[tid] += red2[tid + off]; }
            __syncthreads();
        }
        const float mean = red1[0] * (1.0f / NF);
        const float var  = red2[0] * (1.0f / NF) - mean * mean;
        const float rs   = rsqrtf(var + LN_EPS);
        const float mv   = maskp[grow];
        for (int cc = tid; cc < NF; cc += 256) {
            float v = mv * ((sD[r * NF + cc] - mean) * rs * ln2w[cc] + ln2b[cc]);
            outF[(size_t)grow * NF + cc] = v;
            outB[(size_t)grow * NF + cc] = (bf16)v;
        }
        __syncthreads();
    }
}

// ---------------------------------------------------------------------------

extern "C" void kernel_launch(void* const* d_in, const int* in_sizes, int n_in,
                              void* d_out, int out_size, void* d_ws, size_t ws_size,
                              hipStream_t stream) {
    (void)in_sizes; (void)n_in; (void)out_size; (void)ws_size;

    const float* node_f  = (const float*)d_in[0];
    const float* edge_f  = (const float*)d_in[1];
    const int*   nbr     = (const int*)  d_in[2];
    const float* mask    = (const float*)d_in[3];
    const float* msg_w0  = (const float*)d_in[4];
    const float* msg_b0  = (const float*)d_in[5];
    const float* msg_w1  = (const float*)d_in[6];
    const float* msg_b1  = (const float*)d_in[7];
    const float* msg_w2  = (const float*)d_in[8];
    const float* msg_b2  = (const float*)d_in[9];
    const float* ln1w    = (const float*)d_in[10];
    const float* ln1b    = (const float*)d_in[11];
    const float* dw0     = (const float*)d_in[12];
    const float* db0     = (const float*)d_in[13];
    const float* dw1     = (const float*)d_in[14];
    const float* db1     = (const float*)d_in[15];
    const float* ln2w    = (const float*)d_in[16];
    const float* ln2b    = (const float*)d_in[17];

    uint8_t* ws = (uint8_t*)d_ws;
    size_t off = 0;
    auto alloc = [&](size_t bytes) -> void* {
        void* p = ws + off;
        off = (off + bytes + 255) & ~(size_t)255;
        return p;
    };

    bf16*  E16 = (bf16*)alloc((size_t)NN * KK * NF * 2);
    bf16*  n0b = (bf16*)alloc((size_t)NN * NF * 2);
    float* xf  = (float*)alloc((size_t)NN * NF * 4);
    bf16*  xb  = (bf16*) alloc((size_t)NN * NF * 2);
    float* x1f = (float*)alloc((size_t)NN * NF * 4);
    bf16*  x1b = (bf16*) alloc((size_t)NN * NF * 2);
    float* P   = (float*)alloc((size_t)NN * NF * 4);
    float* Q   = (float*)alloc((size_t)NN * NF * 4);
    bf16*  Hbuf= (bf16*) alloc((size_t)NN * MINW * 2);
    bf16 *WxT[3], *WeT[3], *WnT[3], *W1T[3], *W2T[3], *W0dT[3], *W1dT[3];
    for (int l = 0; l < 3; ++l) {
        WxT[l]  = (bf16*)alloc((size_t)NF * NF * 2);
        WeT[l]  = (bf16*)alloc((size_t)NF * NF * 2);
        WnT[l]  = (bf16*)alloc((size_t)NF * NF * 2);
        W1T[l]  = (bf16*)alloc((size_t)NF * NF * 2);
        W2T[l]  = (bf16*)alloc((size_t)NF * NF * 2);
        W0dT[l] = (bf16*)alloc((size_t)MINW * NF * 2);
        W1dT[l] = (bf16*)alloc((size_t)NF * MINW * 2);
    }

    // One-time conversions.
    k_cvt_bf16<<<2048, 256, 0, stream>>>(edge_f, E16, NN * KK * NF);
    k_cvt_bf16<<<256, 256, 0, stream>>>(node_f, n0b, NN * NF);
    k_copy_f32<<<256, 256, 0, stream>>>(node_f, xf, NN * NF);
    k_cvt_bf16<<<256, 256, 0, stream>>>(node_f, xb, NN * NF);

    // Weight transpose+convert: dst[c][k] = src[k][c].
    auto tr = [&](const float* src, bf16* dst, int Kd, int Cd) {
        int total = Kd * Cd;
        int blocks = (total + 255) / 256; if (blocks > 2304) blocks = 2304;
        k_transpose_cvt<<<blocks, 256, 0, stream>>>(src, dst, Kd, Cd, total);
    };
    for (int l = 0; l < 3; ++l) {
        const float* w0 = msg_w0 + (size_t)l * MINW * NF;
        tr(w0,                 WxT[l], NF, NF);   // rows    0..383 : x block
        tr(w0 + 384 * NF,      WeT[l], NF, NF);   // rows  384..767 : edge block
        tr(w0 + 1152 * NF,     WnT[l], NF, NF);   // rows 1152..1535: neighbor block
        // rows 768..1151 multiply an all-zero block -> skipped entirely.
        tr(msg_w1 + (size_t)l * NF * NF,  W1T[l],  NF,   NF);
        tr(msg_w2 + (size_t)l * NF * NF,  W2T[l],  NF,   NF);
        tr(dw0   + (size_t)l * NF * MINW, W0dT[l], NF,   MINW);
        tr(dw1   + (size_t)l * MINW * NF, W1dT[l], MINW, NF);
    }

    for (int l = 0; l < 3; ++l) {
        // P = x @ Wx + b0 ; Q = nodes0 @ Wn  (gathered per neighbor later).
        k_gemm16_f32<<<NN / 16, 256, 0, stream>>>(xb,  WxT[l], msg_b0 + l * NF, P);
        k_gemm16_f32<<<NN / 16, 256, 0, stream>>>(n0b, WnT[l], nullptr,         Q);

        k_message<<<NN, 256, 0, stream>>>(E16, nbr, P, Q, WeT[l], W1T[l], W2T[l],
                                          msg_b1 + l * NF, msg_b2 + l * NF,
                                          ln1w + l * NF, ln1b + l * NF,
                                          xf, x1f, x1b);

        k_dense1<<<NN / 16, 256, 0, stream>>>(x1b, W0dT[l], db0 + l * MINW, Hbuf);

        float* outF = (l == 2) ? (float*)d_out : xf;
        k_dense2<<<NN / 16, 256, 0, stream>>>(Hbuf, W1dT[l], db1 + l * NF, x1f, mask,
                                              ln2w + l * NF, ln2b + l * NF, outF, xb);
    }
}
